// Quantize_29454885716598
// MI455X (gfx1250) — matrix operand order repaired and checked by reading, not compiled, not run on hardware
//
#include <hip/hip_runtime.h>
#include <hip/hip_bf16.h>

typedef __attribute__((ext_vector_type(16))) __bf16 v16bf;
typedef __attribute__((ext_vector_type(8)))  float  v8f;
typedef __attribute__((ext_vector_type(4)))  unsigned v4u;
typedef __attribute__((ext_vector_type(8)))  int    v8i;
typedef __attribute__((ext_vector_type(4)))  int    v4i;

#define N_TOK 131072
#define N_E   1024
#define E_DIM 64
#define DECAY 0.99f
#define OMD   (1.0f - 0.99f)
#define EPSV  1e-5f

// ---------- bf16 helpers (round-to-nearest-even) ----------
__device__ __forceinline__ unsigned pack2bf(float lo, float hi) {
  unsigned a = __builtin_bit_cast(unsigned, lo);
  unsigned b = __builtin_bit_cast(unsigned, hi);
  a = ((a + 0x7FFFu + ((a >> 16) & 1u)) >> 16) & 0xFFFFu;
  b = ((b + 0x7FFFu + ((b >> 16) & 1u)) >> 16) & 0xFFFFu;
  return a | (b << 16);
}
__device__ __forceinline__ unsigned short f2bf(float f) {
  unsigned a = __builtin_bit_cast(unsigned, f);
  return (unsigned short)(((a + 0x7FFFu + ((a >> 16) & 1u)) >> 16) & 0xFFFFu);
}

// ---------- Kernel A: b_bf[k][d] = bf16(embed[k][d] / ||embed[k]||^2) ----------
__global__ void __launch_bounds__(32) k_embed_scale(const float* __restrict__ embed,
                                                    unsigned short* __restrict__ b_bf) {
  int k = blockIdx.x;           // one code per wave
  int lane = threadIdx.x;       // 0..31
  const float* er = embed + k * E_DIM;
  float v0 = er[lane], v1 = er[lane + 32];
  float s = v0 * v0 + v1 * v1;
  #pragma unroll
  for (int off = 16; off; off >>= 1) s += __shfl_xor(s, off);
  float inv = 1.0f / s;
  unsigned short* outr = b_bf + k * E_DIM;
  outr[lane]      = f2bf(v0 * inv);
  outr[lane + 32] = f2bf(v1 * inv);
}

// ---------- Kernel B: TDM-staged codebook in LDS + bf16 WMMA GEMM + fused argmax ----------
// Block = 8 waves = 128 tokens. Whole scaled codebook (1024x64 bf16 = 128 KB) is DMA'd
// into LDS once per block by the Tensor Data Mover; waves then sweep 64 code tiles,
// 2x v_wmma_f32_16x16x32_bf16 per tile (K=64), B double-buffered from ds_load_b128.
union BU { v16bf v; uint4 q[2]; };

__global__ void __launch_bounds__(256) k_argmax(const float* __restrict__ x,
                                                const unsigned short* __restrict__ b_bf,
                                                int* __restrict__ embed_ind) {
  extern __shared__ unsigned short sB[];   // 1024*64 bf16 = 131072 bytes

  const int lane   = threadIdx.x & 31;
  const int wave   = threadIdx.x >> 5;
  const int tbase  = (blockIdx.x * 8 + wave) * 16;
  const int lane15 = lane & 15;
  const int sel    = lane >> 4;     // 0: lanes 0-15, 1: lanes 16-31

#if __has_builtin(__builtin_amdgcn_tensor_load_to_lds)
  if (wave == 0) {
    // D# group0: count=1, lds_addr=0 (dynamic LDS base), 57b global addr, type=2
    unsigned long long ga = (unsigned long long)b_bf;
    v4u g0 = { 1u, 0u, (unsigned)ga,
               ((unsigned)((ga >> 32) & 0x01FFFFFFull)) | 0x80000000u };
    // D# group1: data_size=3 (8B units), tensor_dim0=16 (128B row), tensor_dim1=1024,
    //            tile_dim0=16, tile_dim1=1024, tensor_dim0_stride=16
    v8i g1 = { (3 << 16),        // workgroup_mask=0 | data_size=8B
               (16 << 16),       // tensor_dim0[15:0] in bits 63:48
               (1024 << 16),     // tensor_dim1[15:0] in bits 95:80
               (16 << 16),       // tile_dim0=16 in bits 127:112
               1024,             // tile_dim1=1024 (bits 143:128), tile_dim2=0
               16,               // tensor_dim0_stride low32
               0, 0 };
    v4i g2 = { 0, 0, 0, 0 };
    v4i g3 = { 0, 0, 0, 0 };
    v8i g4 = { 0, 0, 0, 0, 0, 0, 0, 0 };
    // amdgpu-toolchain (clang-23) 6-arg form: (g0, g1, g2, g3, g4, cpol)
    __builtin_amdgcn_tensor_load_to_lds(g0, g1, g2, g3, g4, 0);
    __builtin_amdgcn_s_wait_tensorcnt(0);
  }
#else
  for (int i = threadIdx.x; i < (N_E * E_DIM * 2) / 16; i += 256)
    ((uint4*)sB)[i] = ((const uint4*)b_bf)[i];
#endif
  __syncthreads();

  // ---- Load + convert A tile (16 tokens x 64 dims) to bf16, contiguous-K A-layout ----
  // per lane: elements 0..15 = K (kbase + 16*sel + 0..15)
  const float* xr = x + (size_t)(tbase + lane15) * E_DIM;
  union { v16bf v; unsigned u[8]; } A0, A1;
  {
    const float* p = xr + 16 * sel;
    #pragma unroll
    for (int i = 0; i < 4; i++) A0.u[i]     = pack2bf(p[2*i],      p[2*i + 1]);
    #pragma unroll
    for (int i = 0; i < 4; i++) A0.u[4 + i] = pack2bf(p[8 + 2*i], p[8 + 2*i + 1]);
    #pragma unroll
    for (int i = 0; i < 4; i++) A1.u[i]     = pack2bf(p[32 + 2*i], p[32 + 2*i + 1]);
    #pragma unroll
    for (int i = 0; i < 4; i++) A1.u[4 + i] = pack2bf(p[40 + 2*i], p[40 + 2*i + 1]);
  }

  float best[8];
  int   bidx[8];
  #pragma unroll
  for (int v = 0; v < 8; v++) { best[v] = -3.4e38f; bidx[v] = 0; }

  // B tile j lives at sB + ((j*16 + lane15) * 64); per lane: 16 contiguous K at 16*sel
  // (uint4 row indices {2*sel, 2*sel+1} for K<32 and {4+2*sel, 5+2*sel} for K>=32).
  BU Ba0, Ba1, Bb0, Bb1;
  {
    const uint4* rowp = (const uint4*)(sB + lane15 * E_DIM);   // tile 0
    Ba0.q[0] = rowp[2 * sel];     Ba0.q[1] = rowp[2 * sel + 1];
    Ba1.q[0] = rowp[4 + 2 * sel]; Ba1.q[1] = rowp[5 + 2 * sel];
  }

  for (int j = 0; j < 64; j += 2) {
    // prefetch tile j+1 into Bb while tile j computes
    {
      const uint4* np = (const uint4*)(sB + (((j + 1) << 4) | lane15) * E_DIM);
      Bb0.q[0] = np[2 * sel];     Bb0.q[1] = np[2 * sel + 1];
      Bb1.q[0] = np[4 + 2 * sel]; Bb1.q[1] = np[5 + 2 * sel];
    }
    {
      const int code = (j << 4) | lane15;
      v8f c = {0.f, 0.f, 0.f, 0.f, 0.f, 0.f, 0.f, 0.f};
      c = __builtin_amdgcn_wmma_f32_16x16x32_bf16(false, A0.v, false, Ba0.v, (short)0, c, false, false);
      c = __builtin_amdgcn_wmma_f32_16x16x32_bf16(false, A1.v, false, Ba1.v, (short)0, c, false, false);
      #pragma unroll
      for (int v = 0; v < 8; v++) {
        bool take = c[v] > best[v];
        best[v] = take ? c[v] : best[v];
        bidx[v] = take ? code : bidx[v];
      }
    }
    // prefetch tile j+2 into Ba while tile j+1 computes (uniform guard, EXEC untouched)
    if (j + 2 < 64) {
      const uint4* np = (const uint4*)(sB + (((j + 2) << 4) | lane15) * E_DIM);
      Ba0.q[0] = np[2 * sel];     Ba0.q[1] = np[2 * sel + 1];
      Ba1.q[0] = np[4 + 2 * sel]; Ba1.q[1] = np[5 + 2 * sel];
    }
    {
      const int code = ((j + 1) << 4) | lane15;
      v8f c = {0.f, 0.f, 0.f, 0.f, 0.f, 0.f, 0.f, 0.f};
      c = __builtin_amdgcn_wmma_f32_16x16x32_bf16(false, A0.v, false, Bb0.v, (short)0, c, false, false);
      c = __builtin_amdgcn_wmma_f32_16x16x32_bf16(false, A1.v, false, Bb1.v, (short)0, c, false, false);
      #pragma unroll
      for (int v = 0; v < 8; v++) {
        bool take = c[v] > best[v];
        best[v] = take ? c[v] : best[v];
        bidx[v] = take ? code : bidx[v];
      }
    }
  }

  // ---- cross-lane argmax within each 16-lane half (tokens v / v+8), first-max tiebreak ----
  #pragma unroll
  for (int off = 8; off; off >>= 1) {
    #pragma unroll
    for (int v = 0; v < 8; v++) {
      float ov = __shfl_xor(best[v], off);
      int   oi = __shfl_xor(bidx[v], off);
      bool take = (ov > best[v]) || (ov == best[v] && oi < bidx[v]);
      best[v] = take ? ov : best[v];
      bidx[v] = take ? oi : bidx[v];
    }
  }
  if (lane15 == 0) {
    int tb = tbase + sel * 8;
    #pragma unroll
    for (int v = 0; v < 8; v++) embed_ind[tb + v] = bidx[v];
  }
}

// ---------- Kernel C: segment stats + diff accumulation ----------
__global__ void __launch_bounds__(256) k_stats(const float* __restrict__ x,
                                               const float* __restrict__ embed,
                                               const int* __restrict__ embed_ind,
                                               float* __restrict__ counts,
                                               float* __restrict__ embed_sum,
                                               float* __restrict__ diff_acc) {
  int n = blockIdx.x * 256 + threadIdx.x;
  int ind = embed_ind[n];
  atomicAdd(&counts[ind], 1.0f);
  const float* xr = x + (size_t)n * E_DIM;
  const float* er = embed + (size_t)ind * E_DIM;
  float* es = embed_sum + (size_t)ind * E_DIM;
  float dsum = 0.f;
  #pragma unroll 4
  for (int d = 0; d < E_DIM; d++) {
    float xv = xr[d];
    atomicAdd(&es[d], xv);
    float dd = er[d] - xv;
    dsum += dd * dd;
  }
  #pragma unroll
  for (int off = 16; off; off >>= 1) dsum += __shfl_xor(dsum, off);
  if ((threadIdx.x & 31) == 0) atomicAdd(diff_acc, dsum);
}

// ---------- Kernel D: EMA update + total-n reduction ----------
__global__ void __launch_bounds__(256) k_ema(const float* __restrict__ cluster_size,
                                             const float* __restrict__ embed_to_avg,
                                             const float* __restrict__ counts,
                                             const float* __restrict__ embed_sum,
                                             float* __restrict__ out_ncs,
                                             float* __restrict__ out_nea,
                                             float* __restrict__ n_acc) {
  int t = blockIdx.x * 256 + threadIdx.x;   // 0..65535
  out_nea[t] = embed_to_avg[t] * DECAY + OMD * embed_sum[t];
  if ((t & 63) == 0) {
    int k = t >> 6;
    float ncs = cluster_size[k] * DECAY + OMD * counts[k];
    out_ncs[k] = ncs;
    atomicAdd(n_acc, ncs);
  }
}

// ---------- Kernel E: finalize new_embed + diff ----------
__global__ void __launch_bounds__(256) k_final(const float* __restrict__ out_ncs,
                                               const float* __restrict__ out_nea,
                                               const float* __restrict__ n_acc,
                                               const float* __restrict__ diff_acc,
                                               float* __restrict__ out_embed,
                                               float* __restrict__ out_diff) {
  int t = blockIdx.x * 256 + threadIdx.x;   // 0..65535
  float n = *n_acc;
  int k = t >> 6;
  float cs = (out_ncs[k] + EPSV) / (n + (float)N_E * EPSV) * n;
  out_embed[t] = out_nea[t] / cs;
  if (t == 0) *out_diff = *diff_acc * (1.0f / 8388608.0f);  // / (N_TOK*E_DIM)
}

extern "C" void kernel_launch(void* const* d_in, const int* in_sizes, int n_in,
                              void* d_out, int out_size, void* d_ws, size_t ws_size,
                              hipStream_t stream) {
  (void)in_sizes; (void)n_in; (void)out_size; (void)ws_size;
  const float* x            = (const float*)d_in[0];
  const float* embed        = (const float*)d_in[1];
  const float* cluster_size = (const float*)d_in[2];
  const float* embed_to_avg = (const float*)d_in[3];

  float* out       = (float*)d_out;
  float* out_diff  = out;                       // [1]
  float* out_embed = out + 1;                   // [1024*64]
  float* out_ncs   = out + 1 + 65536;           // [1024]
  float* out_nea   = out + 1 + 65536 + 1024;    // [1024*64]

  char* wsb = (char*)d_ws;
  unsigned short* b_bf      = (unsigned short*)(wsb);           // 131072 B
  int*            embed_ind = (int*)(wsb + 131072);             // 524288 B
  float*          counts    = (float*)(wsb + 655360);           // 4096 B
  float*          embed_sum = (float*)(wsb + 659456);           // 262144 B
  float*          diff_acc  = (float*)(wsb + 921600);           // 4 B
  float*          n_acc     = (float*)(wsb + 921604);           // 4 B

  // zero counts + embed_sum + diff_acc + n_acc (capture-legal)
  (void)hipMemsetAsync(wsb + 655360, 0, 266248, stream);

  k_embed_scale<<<N_E, 32, 0, stream>>>(embed, b_bf);
  k_argmax<<<N_TOK / (16 * 8), 256, N_E * E_DIM * 2 /*128 KB dyn LDS*/, stream>>>(x, b_bf, embed_ind);
  k_stats<<<N_TOK / 256, 256, 0, stream>>>(x, embed, embed_ind, counts, embed_sum, diff_acc);
  k_ema<<<65536 / 256, 256, 0, stream>>>(cluster_size, embed_to_avg, counts, embed_sum,
                                         out_ncs, out_nea, n_acc);
  k_final<<<65536 / 256, 256, 0, stream>>>(out_ncs, out_nea, n_acc, diff_acc,
                                           out_embed, out_diff);
}
